// GraphAttention_87222195847390
// MI455X (gfx1250) — matrix operand order repaired and checked by reading, not compiled, not run on hardware
//
#include <hip/hip_runtime.h>

// ---------------------------------------------------------------------------
// Graph-biased causal multi-head attention for MI455X (gfx1250, wave32).
// bf16 WMMA (v_wmma_f32_16x16x32_bf16) for all three matmul stages,
// flash-attention with online softmax for the middle stage.
// Round 4: all streaming loads expressed as uniform-base + lane-invariant
// 32-bit offset so the k-advance stays in SALU (saddr-form loads, invariant
// offset VGPRs -> no per-iteration v_add64 and no WMMA->VALU WAR nops);
// ping-pong pipeline restructured as prologue / steady loop / tail.
// ---------------------------------------------------------------------------

typedef __attribute__((ext_vector_type(16))) __bf16       v16bf;
typedef __attribute__((ext_vector_type(8)))  float        v8f;
typedef __attribute__((ext_vector_type(4)))  unsigned int v4u;

#define LL     512          // sequence length
#define BB     16           // batch
#define EE     512          // embed dim
#define HH     8            // heads
#define HD     64           // head dim
#define BH     128          // B*H
#define MROWS  8192         // L*B
#define NQKV   1536         // 3*E

#define WMMA_BF16(A, B, C) \
  __builtin_amdgcn_wmma_f32_16x16x32_bf16(false, (A), false, (B), (short)0, (C), false, false)

static __device__ __forceinline__ unsigned short f2bf(float f) {
  unsigned int u = __builtin_bit_cast(unsigned int, f);
  u += 0x7FFFu + ((u >> 16) & 1u);                 // round-to-nearest-even
  return (unsigned short)(u >> 16);
}

union FragU {
  v16bf bf;
  v4u   q[2];
  unsigned short us[16];
};

// fragment load: uniform base pointer + per-lane (loop-invariant) offsets
static __device__ __forceinline__ v16bf ld_frag(const unsigned short* base,
                                                int o0, int o1) {
  FragU f;
  f.q[0] = *(const v4u*)(base + o0);
  f.q[1] = *(const v4u*)(base + o1);
  return f.bf;
}

static __device__ __forceinline__ v8f vzero8() {
  v8f z = {0.f, 0.f, 0.f, 0.f, 0.f, 0.f, 0.f, 0.f};
  return z;
}

// ---------------------------------------------------------------------------
// Kernel 0: f32 -> bf16 conversion
// ---------------------------------------------------------------------------
__global__ __launch_bounds__(256) void k_cvt_bf16(const float* __restrict__ src,
                                                  unsigned short* __restrict__ dst,
                                                  int n) {
  int i = blockIdx.x * 256 + threadIdx.x;
  if (i < n) dst[i] = f2bf(src[i]);
}

// ---------------------------------------------------------------------------
// GEMM core: 32x64 wave tile, K = EE in steps of 32, ping-pong x2 unroll.
//   abase/bbase : uniform base pointers (advanced by the uniform k index)
//   aoff0/aoff1 : per-lane A row offsets (ushort units), loop-invariant
//   boff[4]     : per-lane B row offsets, loop-invariant
// ---------------------------------------------------------------------------
static __device__ __forceinline__ void gemm_32x64(const unsigned short* abase,
                                                  int aoff0, int aoff1,
                                                  const unsigned short* bbase,
                                                  const int boff[4],
                                                  v8f acc[2][4]) {
  // prologue: set A <- k = 0
  v16bf a0A = ld_frag(abase, aoff0, aoff0 + 8);
  v16bf a1A = ld_frag(abase, aoff1, aoff1 + 8);
  v16bf bA[4];
  #pragma unroll
  for (int j = 0; j < 4; ++j) bA[j] = ld_frag(bbase, boff[j], boff[j] + 8);

  for (int k0 = 0; k0 < EE - 64; k0 += 64) {
    // set B <- k0+32
    const unsigned short* aB = abase + k0 + 32;   // uniform
    const unsigned short* bB = bbase + k0 + 32;   // uniform
    v16bf a0B = ld_frag(aB, aoff0, aoff0 + 8);
    v16bf a1B = ld_frag(aB, aoff1, aoff1 + 8);
    v16bf bBf[4];
    #pragma unroll
    for (int j = 0; j < 4; ++j) bBf[j] = ld_frag(bB, boff[j], boff[j] + 8);

    // compute with set A (k0)
    #pragma unroll
    for (int j = 0; j < 4; ++j) {
      acc[0][j] = WMMA_BF16(a0A, bA[j], acc[0][j]);
      acc[1][j] = WMMA_BF16(a1A, bA[j], acc[1][j]);
    }

    // set A <- k0+64
    const unsigned short* aN = abase + k0 + 64;   // uniform
    const unsigned short* bN = bbase + k0 + 64;   // uniform
    a0A = ld_frag(aN, aoff0, aoff0 + 8);
    a1A = ld_frag(aN, aoff1, aoff1 + 8);
    #pragma unroll
    for (int j = 0; j < 4; ++j) bA[j] = ld_frag(bN, boff[j], boff[j] + 8);

    // compute with set B (k0+32)
    #pragma unroll
    for (int j = 0; j < 4; ++j) {
      acc[0][j] = WMMA_BF16(a0B, bBf[j], acc[0][j]);
      acc[1][j] = WMMA_BF16(a1B, bBf[j], acc[1][j]);
    }
  }

  // tail: set A holds k = EE-64; load set B at EE-32, compute both
  {
    const unsigned short* aB = abase + EE - 32;
    const unsigned short* bB = bbase + EE - 32;
    v16bf a0B = ld_frag(aB, aoff0, aoff0 + 8);
    v16bf a1B = ld_frag(aB, aoff1, aoff1 + 8);
    v16bf bBf[4];
    #pragma unroll
    for (int j = 0; j < 4; ++j) bBf[j] = ld_frag(bB, boff[j], boff[j] + 8);

    #pragma unroll
    for (int j = 0; j < 4; ++j) {
      acc[0][j] = WMMA_BF16(a0A, bA[j], acc[0][j]);
      acc[1][j] = WMMA_BF16(a1A, bA[j], acc[1][j]);
    }
    #pragma unroll
    for (int j = 0; j < 4; ++j) {
      acc[0][j] = WMMA_BF16(a0B, bBf[j], acc[0][j]);
      acc[1][j] = WMMA_BF16(a1B, bBf[j], acc[1][j]);
    }
  }
}

// ---------------------------------------------------------------------------
// Kernel 1: QKV projection  qkv = x @ Wqkv^T + b, routed into head layouts.
//   xb : (MROWS, EE) bf16      wb : (NQKV, EE) bf16 (row n contiguous in k)
//   Wq : (bh, l, d) bf16, pre-scaled by 1/sqrt(hd)
//   Wk : (bh, l, d) bf16
//   Vt : (bh, d, l) bf16 (transposed so attn@V B-operand is contiguous)
// Grid = (NQKV/64, MROWS/256), block = 8 waves, wave tile 32x64.
// ---------------------------------------------------------------------------
__global__ __launch_bounds__(256) void k_qkv(const unsigned short* __restrict__ xb,
                                             const unsigned short* __restrict__ wb,
                                             const float* __restrict__ bias,
                                             unsigned short* __restrict__ Wq,
                                             unsigned short* __restrict__ Wk,
                                             unsigned short* __restrict__ Vt) {
  const int w    = threadIdx.x >> 5;
  const int lane = threadIdx.x & 31;
  const int lid  = lane & 15;
  const int half = lane >> 4;
  const int m0   = blockIdx.y * 256 + w * 32;
  const int n0   = blockIdx.x * 64;

  v8f acc[2][4];
  #pragma unroll
  for (int g = 0; g < 2; ++g)
    #pragma unroll
    for (int j = 0; j < 4; ++j) acc[g][j] = vzero8();

  const int aoff0 = (m0 + lid) * EE + half * 16;
  const int aoff1 = aoff0 + 16 * EE;
  int boff[4];
  #pragma unroll
  for (int j = 0; j < 4; ++j) boff[j] = (n0 + 16 * j + lid) * EE + half * 16;

  gemm_32x64(xb, aoff0, aoff1, wb, boff, acc);

  // epilogue: the 64-col tile lies entirely in one (section, head)
  const int sect = n0 / EE;             // 0=q 1=k 2=v
  const int h    = (n0 % EE) / HD;
  #pragma unroll
  for (int g = 0; g < 2; ++g)
    #pragma unroll
    for (int j = 0; j < 4; ++j) {
      const int n  = n0 + 16 * j + lid;
      const int d  = 16 * j + lid;      // dim within head
      const float bi = bias[n];
      #pragma unroll
      for (int r = 0; r < 8; ++r) {
        const int m  = m0 + g * 16 + r + 8 * half;  // row in C layout
        const int l  = m >> 4;          // m / B
        const int bb = m & 15;          // m % B
        const int bh = bb * HH + h;
        const float v = acc[g][j][r] + bi;
        if (sect == 0)
          Wq[((size_t)bh * LL + l) * HD + d] = f2bf(v * 0.125f); // 1/sqrt(64)
        else if (sect == 1)
          Wk[((size_t)bh * LL + l) * HD + d] = f2bf(v);
        else
          Vt[((size_t)bh * HD + d) * LL + l] = f2bf(v);
      }
    }
}

// ---------------------------------------------------------------------------
// Kernel 2: flash attention with causal mask, key-padding and edge bias.
// Grid = BH * (L/128); block 256 = 8 waves; each wave owns 16 query rows and
// streams 32-key steps with online softmax. K/V loads use uniform-base(kt) +
// invariant lane offsets; V loads issue early so their latency hides under
// the score WMMAs + softmax VALU. P transposed via per-wave LDS tile.
// ---------------------------------------------------------------------------
__global__ __launch_bounds__(256) void k_attn(const unsigned short* __restrict__ Wq,
                                              const unsigned short* __restrict__ Wk,
                                              const unsigned short* __restrict__ Vt,
                                              const int* __restrict__ edge,
                                              const unsigned char* __restrict__ kpm,
                                              const float* __restrict__ edge_emb,
                                              unsigned short* __restrict__ ao) {
  __shared__ float eb[16 * HH];
  __shared__ __attribute__((aligned(16))) unsigned short pbuf[8][16][32];

  const int t = threadIdx.x;
  if (t < 16 * HH) eb[t] = edge_emb[t];
  __syncthreads();

  const int w    = t >> 5;
  const int lane = t & 31;
  const int lid  = lane & 15;
  const int half = lane >> 4;
  const int bh   = blockIdx.x & (BH - 1);
  const int qb   = blockIdx.x >> 7;
  const int bb   = bh >> 3;             // batch
  const int h    = bh & 7;              // head
  const int q0   = qb * 128 + w * 16;

  const unsigned short* Qb = Wq + (size_t)bh * LL * HD;
  const unsigned short* Kb = Wk + (size_t)bh * LL * HD;
  const unsigned short* Vb = Vt + (size_t)bh * HD * LL;

  // Q fragments (resident for the whole key loop): kk covers hd 0-31 / 32-63
  v16bf qf[2];
  {
    const int qo = (q0 + lid) * HD + half * 16;
    qf[0] = ld_frag(Qb, qo,      qo + 8);
    qf[1] = ld_frag(Qb, qo + 32, qo + 40);
  }

  // loop-invariant per-lane offsets (kt advances the uniform base instead)
  int koff[2], voff[4];
  #pragma unroll
  for (int j = 0; j < 2; ++j) koff[j] = (16 * j + lid) * HD + half * 16;
  #pragma unroll
  for (int j = 0; j < 4; ++j) voff[j] = (16 * j + lid) * LL + half * 16;

  float mi[8], li[8];
  v8f o[4];
  #pragma unroll
  for (int r = 0; r < 8; ++r) { mi[r] = -1e30f; li[r] = 0.f; }
  #pragma unroll
  for (int j = 0; j < 4; ++j) o[j] = vzero8();

  const int kend = q0 + 16;             // causal: keys <= q0+15
  for (int kt = 0; kt < kend; kt += 32) {
    const unsigned short* Kt = Kb + kt * HD;   // uniform
    const unsigned short* Vtt = Vb + kt;       // uniform

    // ---- K fragments (2 tiles x 2 hd-steps) and V fragments (early) ----
    v16bf kf[2][2];
    #pragma unroll
    for (int j = 0; j < 2; ++j) {
      kf[j][0] = ld_frag(Kt, koff[j],      koff[j] + 8);
      kf[j][1] = ld_frag(Kt, koff[j] + 32, koff[j] + 40);
    }
    v16bf vf[4];
    #pragma unroll
    for (int j = 0; j < 4; ++j) vf[j] = ld_frag(Vtt, voff[j], voff[j] + 8);

    // ---- scores: two 16x16 tiles over 32 keys -------------------------
    v8f s[2];
    #pragma unroll
    for (int j = 0; j < 2; ++j) {
      v8f sa = vzero8();
      sa = WMMA_BF16(qf[0], kf[j][0], sa);
      sa = WMMA_BF16(qf[1], kf[j][1], sa);
      s[j] = sa;
    }

    // ---- mask + edge bias, per-row max --------------------------------
    float rowmax[8];
    #pragma unroll
    for (int r = 0; r < 8; ++r) rowmax[r] = -1e30f;
    #pragma unroll
    for (int j = 0; j < 2; ++j) {
      const int key = kt + 16 * j + lid;
      const bool kp = kpm[bb * LL + key] != 0;
      #pragma unroll
      for (int r = 0; r < 8; ++r) {
        const int q = q0 + r + 8 * half;
        float v = s[j][r];
        if (kp || key > q) {
          v = -1e30f;
        } else {
          const int e = edge[((size_t)bb * LL + q) * LL + key];
          v += eb[e * HH + h];
        }
        s[j][r] = v;
        rowmax[r] = fmaxf(rowmax[r], v);
      }
    }
    #pragma unroll
    for (int r = 0; r < 8; ++r) {       // reduce across the 16 lanes of a row
      float v = rowmax[r];
      v = fmaxf(v, __shfl_xor(v, 1, 32));
      v = fmaxf(v, __shfl_xor(v, 2, 32));
      v = fmaxf(v, __shfl_xor(v, 4, 32));
      v = fmaxf(v, __shfl_xor(v, 8, 32));
      rowmax[r] = v;
    }

    // ---- online softmax rescale ---------------------------------------
    float scale[8];
    #pragma unroll
    for (int r = 0; r < 8; ++r) {
      const float mnew = fmaxf(mi[r], rowmax[r]);
      const float sc   = __expf(mi[r] - mnew);
      mi[r] = mnew;
      li[r] *= sc;
      scale[r] = sc;
    }
    #pragma unroll
    for (int j = 0; j < 4; ++j)
      #pragma unroll
      for (int r = 0; r < 8; ++r) o[j][r] = o[j][r] * scale[r];

    // ---- P = exp(s - m), row sums, transpose via LDS ------------------
    float rsum[8];
    #pragma unroll
    for (int r = 0; r < 8; ++r) rsum[r] = 0.f;
    #pragma unroll
    for (int j = 0; j < 2; ++j)
      #pragma unroll
      for (int r = 0; r < 8; ++r) {
        const float p = __expf(s[j][r] - mi[r]);
        rsum[r] += p;
        pbuf[w][r + 8 * half][16 * j + lid] = f2bf(p);
      }
    #pragma unroll
    for (int r = 0; r < 8; ++r) {
      float v = rsum[r];
      v += __shfl_xor(v, 1, 32);
      v += __shfl_xor(v, 2, 32);
      v += __shfl_xor(v, 4, 32);
      v += __shfl_xor(v, 8, 32);
      li[r] += v;
    }
    __builtin_amdgcn_wave_barrier();    // LDS per-wave, in-order; fence codegen

    // ---- P fragment (A layout) + 4 WMMA against V^T -------------------
    v16bf pf;
    {
      const unsigned short* pr = &pbuf[w][lid][0];
      pf = ld_frag(pr, half * 16, half * 16 + 8);
    }
    #pragma unroll
    for (int j = 0; j < 4; ++j) o[j] = WMMA_BF16(pf, vf[j], o[j]);
    __builtin_amdgcn_wave_barrier();    // keep next-iter LDS stores behind reads
  }

  // ---- normalize and write (l, b, E) bf16 -----------------------------
  #pragma unroll
  for (int j = 0; j < 4; ++j)
    #pragma unroll
    for (int r = 0; r < 8; ++r) {
      const int q = q0 + r + 8 * half;
      const float v = o[j][r] / li[r];
      ao[((size_t)q * BB + bb) * EE + h * HD + 16 * j + lid] = f2bf(v);
    }
}

// ---------------------------------------------------------------------------
// Kernel 3: output projection  out = ao @ Wout^T + b   (f32 result)
// ---------------------------------------------------------------------------
__global__ __launch_bounds__(256) void k_outproj(const unsigned short* __restrict__ ab,
                                                 const unsigned short* __restrict__ wo,
                                                 const float* __restrict__ bias,
                                                 float* __restrict__ out) {
  const int w    = threadIdx.x >> 5;
  const int lane = threadIdx.x & 31;
  const int lid  = lane & 15;
  const int half = lane >> 4;
  const int m0   = blockIdx.y * 256 + w * 32;
  const int n0   = blockIdx.x * 64;

  v8f acc[2][4];
  #pragma unroll
  for (int g = 0; g < 2; ++g)
    #pragma unroll
    for (int j = 0; j < 4; ++j) acc[g][j] = vzero8();

  const int aoff0 = (m0 + lid) * EE + half * 16;
  const int aoff1 = aoff0 + 16 * EE;
  int boff[4];
  #pragma unroll
  for (int j = 0; j < 4; ++j) boff[j] = (n0 + 16 * j + lid) * EE + half * 16;

  gemm_32x64(ab, aoff0, aoff1, wo, boff, acc);

  #pragma unroll
  for (int g = 0; g < 2; ++g)
    #pragma unroll
    for (int j = 0; j < 4; ++j) {
      const int n = n0 + 16 * j + lid;
      const float bi = bias[n];
      #pragma unroll
      for (int r = 0; r < 8; ++r) {
        const int m = m0 + g * 16 + r + 8 * half;
        out[(size_t)m * EE + n] = acc[g][j][r] + bi;
      }
    }
}

// ---------------------------------------------------------------------------
extern "C" void kernel_launch(void* const* d_in, const int* in_sizes, int n_in,
                              void* d_out, int out_size, void* d_ws, size_t ws_size,
                              hipStream_t stream) {
  (void)in_sizes; (void)n_in; (void)out_size; (void)ws_size;

  const float*         x    = (const float*)d_in[0];
  const int*           edge = (const int*)d_in[1];
  const unsigned char* kpm  = (const unsigned char*)d_in[2];
  // d_in[3] = attn_mask: reference builds a causal mask; applied analytically.
  const float* ipw  = (const float*)d_in[4];
  const float* ipb  = (const float*)d_in[5];
  const float* opw  = (const float*)d_in[6];
  const float* opb  = (const float*)d_in[7];
  const float* eemb = (const float*)d_in[8];
  float* out = (float*)d_out;

  // workspace layout (bf16 buffers), ~42 MB total
  unsigned short* xb  = (unsigned short*)d_ws;
  unsigned short* wib = xb  + (size_t)MROWS * EE;       //  8 MB
  unsigned short* wob = wib + (size_t)NQKV * EE;        // +1.5 MB
  unsigned short* Wq  = wob + (size_t)EE * EE;          // +0.5 MB
  unsigned short* Wk  = Wq  + (size_t)BH * LL * HD;     // +8 MB
  unsigned short* Vt  = Wk  + (size_t)BH * LL * HD;     // +8 MB
  unsigned short* ao  = Vt  + (size_t)BH * LL * HD;     // +8 MB (+8 MB)

  k_cvt_bf16<<<(MROWS * EE + 255) / 256, 256, 0, stream>>>(x, xb, MROWS * EE);
  k_cvt_bf16<<<(NQKV * EE + 255) / 256, 256, 0, stream>>>(ipw, wib, NQKV * EE);
  k_cvt_bf16<<<(EE * EE + 255) / 256, 256, 0, stream>>>(opw, wob, EE * EE);

  k_qkv<<<dim3(NQKV / 64, MROWS / 256), 256, 0, stream>>>(xb, wib, ipb, Wq, Wk, Vt);
  k_attn<<<BH * (LL / 128), 256, 0, stream>>>(Wq, Wk, Vt, edge, kpm, eemb, ao);
  k_outproj<<<dim3(EE / 64, MROWS / 256), 256, 0, stream>>>(ao, wob, opb, out);
}
